// AdditiveAttention_10007273799882
// MI455X (gfx1250) — matrix operand order repaired and checked by reading, not compiled, not run on hardware
//
#include <hip/hip_runtime.h>
#include <hip/hip_bf16.h>
#include <stdint.h>

// ---------------------------------------------------------------------------
// AdditiveAttention for MI455X (gfx1250, wave32, WMMA f32_16x16x32_f16)
//
// B=8, S=4096, D=1024, H=16, Dh=64.  M = B*S = 32768.
// 4 big GEMMs [M,1024]x[1024,1024] (~275 GFLOP) on the f16 WMMA path with
// f32 accumulate.  p/u are never materialized: folded into per-batch
// modified weights (wk*qg, out_w*kg), saving ~0.5 GB of HBM traffic.
// GEMM waves compute 32x64 tiles; the K loop is 2x-unrolled with ping-pong
// fragment buffers (no cur<-nxt copies) so loads for K-step k+32 overlap the
// WMMAs of K-step k with zero v_mov overhead.
// ---------------------------------------------------------------------------

typedef __attribute__((ext_vector_type(16))) _Float16 v16h;
typedef __attribute__((ext_vector_type(8)))  _Float16 v8h;
typedef __attribute__((ext_vector_type(4)))  _Float16 v4h;
typedef __attribute__((ext_vector_type(8)))  float    v8f;

#define DTOK 1024
#define NH   16
#define DH   64
#define BB   8
#define SS   4096
#define MM   (BB * SS)
#define TCH  8            // token chunks for pooling reduction
#define GVEC (BB * DTOK)  // 8192 elements in qg/kg

// ---------------------------------------------------------------------------
__global__ __launch_bounds__(256) void cvt_f32_to_f16(
    const float* __restrict__ in, _Float16* __restrict__ out, long n4) {
  long i = (long)blockIdx.x * 256 + threadIdx.x;
  if (i >= n4) return;
  float4 v = ((const float4*)in)[i];
  v4h o;
  o[0] = (_Float16)v.x; o[1] = (_Float16)v.y;
  o[2] = (_Float16)v.z; o[3] = (_Float16)v.w;
  ((v4h*)out)[i] = o;
}

// A-fragment: lane (Li,hi) holds K = {16hi..16hi+15}
__device__ __forceinline__ v16h load_a_frag(const _Float16* aRow, int k0,
                                            int hi) {
  return *(const v16h*)(aRow + k0 + 16 * hi);
}

template <int MT, int NT>
__device__ __forceinline__ void load_frags(const _Float16* const (&aRow)[MT],
                                           const _Float16* const (&bRow)[NT],
                                           int k0, int hi, v16h (&a)[MT],
                                           v16h (&b)[NT]) {
#pragma unroll
  for (int mt = 0; mt < MT; ++mt) a[mt] = load_a_frag(aRow[mt], k0, hi);
#pragma unroll
  for (int nt = 0; nt < NT; ++nt) b[nt] = *(const v16h*)(bRow[nt] + k0);
}

template <int MT, int NT>
__device__ __forceinline__ void wmma_step(v8f (&acc)[MT][NT],
                                          const v16h (&a)[MT],
                                          const v16h (&b)[NT]) {
#pragma unroll
  for (int mt = 0; mt < MT; ++mt)
#pragma unroll
    for (int nt = 0; nt < NT; ++nt)
      acc[mt][nt] = __builtin_amdgcn_wmma_f32_16x16x32_f16(
          false, a[mt], false, b[nt], (short)0, acc[mt][nt], false, false);
}

// ---------------------------------------------------------------------------
// WMMA GEMM:  C[b][m][n] = (sum_k A[b][m][k]*Bw[b][n][k] + bias[n])*scale
//             (+ addh[idx])     A,Bw f16 row-major, ld == K.
// Wave tile: (MT*16) x (NT*16).  8 waves arranged WM x WN per block.
// Requires: K % 64 == 0, M % (WM*MT*16) == 0, N % (WN*NT*16) == 0.
// ---------------------------------------------------------------------------
template <int MT, int NT, int WM, int WN, bool HAS_BIAS, bool HAS_ADD,
          bool WRITE_F32, bool WRITE_F16>
__global__ __launch_bounds__(256) void gemm_wmma_f16(
    const _Float16* __restrict__ A,     // [batch][M][K]
    const _Float16* __restrict__ Bw,    // [batch][N][K] (torch [out,in])
    const float*    __restrict__ bias,  // [N] or null
    const _Float16* __restrict__ addh,  // f16 addend indexed like C
    float*          __restrict__ Cf,    // f32 out
    _Float16*       __restrict__ Ch,    // f16 out
    int K, int ldc,
    long strideA, long strideB, long strideC, float scale) {
  const int lane = threadIdx.x & 31;
  const int wave = threadIdx.x >> 5;  // 0..7 (WM*WN == 8)
  const int wm = wave % WM, wn = wave / WM;
  const int Li = lane & 15;           // A-row / B,C-column within tile
  const int hi = lane >> 4;
  const int bz = blockIdx.z;

  const long m0 = ((long)blockIdx.x * WM + wm) * (MT * 16);
  const long n0 = ((long)blockIdx.y * WN + wn) * (NT * 16);

  const _Float16* Ab = A  + (long)bz * strideA;
  const _Float16* Bb = Bw + (long)bz * strideB;

  const _Float16* aRow[MT];
#pragma unroll
  for (int mt = 0; mt < MT; ++mt)
    aRow[mt] = Ab + (m0 + mt * 16 + Li) * (long)K;
  // B: lane (Li,hi) holds 16 contiguous K at offset 16*hi of row n0+nt*16+Li
  const _Float16* bRow[NT];
#pragma unroll
  for (int nt = 0; nt < NT; ++nt)
    bRow[nt] = Bb + (n0 + nt * 16 + Li) * (long)K + 16 * hi;

  v8f acc[MT][NT];
#pragma unroll
  for (int mt = 0; mt < MT; ++mt)
#pragma unroll
    for (int nt = 0; nt < NT; ++nt)
#pragma unroll
      for (int r = 0; r < 8; ++r) acc[mt][nt][r] = 0.0f;

  // ---- ping-pong pipelined K loop (K/32 steps, even count) ----
  v16h a0[MT], b0[NT], a1[MT], b1[NT];
  load_frags<MT, NT>(aRow, bRow, 0, hi, a0, b0);
  int k0 = 32;
  const int halfIters = K / 64 - 1;
  for (int it = 0; it < halfIters; ++it) {
    load_frags<MT, NT>(aRow, bRow, k0, hi, a1, b1);
    wmma_step<MT, NT>(acc, a0, b0);
    load_frags<MT, NT>(aRow, bRow, k0 + 32, hi, a0, b0);
    wmma_step<MT, NT>(acc, a1, b1);
    k0 += 64;
  }
  load_frags<MT, NT>(aRow, bRow, k0, hi, a1, b1);
  wmma_step<MT, NT>(acc, a0, b0);
  wmma_step<MT, NT>(acc, a1, b1);

  // ---- store.  C layout: lane L -> col L&15, rows (L>>4)*8 + r ----
#pragma unroll
  for (int mt = 0; mt < MT; ++mt) {
#pragma unroll
    for (int nt = 0; nt < NT; ++nt) {
      const long col = n0 + nt * 16 + Li;
      const float bv = HAS_BIAS ? bias[col] : 0.0f;
#pragma unroll
      for (int r = 0; r < 8; ++r) {
        const long row = m0 + mt * 16 + hi * 8 + r;
        const long idx = (long)bz * strideC + row * (long)ldc + col;
        float v = (acc[mt][nt][r] + bv) * scale;
        if (HAS_ADD)   v += (float)addh[idx];
        if (WRITE_F32) Cf[idx] = v;
        if (WRITE_F16) Ch[idx] = (_Float16)v;
      }
    }
  }
}

// ---------------------------------------------------------------------------
// Softmax over tokens: x[b][t][h], softmax over t per (b,h); one block per b.
// ---------------------------------------------------------------------------
__global__ __launch_bounds__(256) void softmax_tokens(float* __restrict__ x,
                                                      int S) {
  __shared__ float red[256];
  __shared__ float stat[16];
  const int b = blockIdx.x, tid = threadIdx.x;
  const int h = tid & 15, tg = tid >> 4;
  float* base = x + (long)b * S * NH;

  float m = -3.0e38f;
  for (int t = tg; t < S; t += 16) m = fmaxf(m, base[t * NH + h]);
  red[tid] = m;
  __syncthreads();
  if (tid < 16) {
    float mm = red[tid];
    for (int g = 1; g < 16; ++g) mm = fmaxf(mm, red[g * 16 + tid]);
    stat[tid] = mm;
  }
  __syncthreads();
  const float mx = stat[h];

  float s = 0.0f;
  for (int t = tg; t < S; t += 16) s += __expf(base[t * NH + h] - mx);
  red[tid] = s;
  __syncthreads();
  if (tid < 16) {
    float ss = 0.0f;
    for (int g = 0; g < 16; ++g) ss += red[g * 16 + tid];
    stat[tid] = 1.0f / ss;
  }
  __syncthreads();
  const float inv = stat[h];
  for (int t = tg; t < S; t += 16) {
    const int i = t * NH + h;
    base[i] = __expf(base[i] - mx) * inv;
  }
}

// ---------------------------------------------------------------------------
// Pooling, stage 1 (partials): part[c][b*1024+h*64+d] =
//   sum_{t in chunk c} w[b][t][h] * X[b][t][h*64+d]
// grid (H, B, TCH), block 64.  Deterministic two-stage (no float atomics).
// ---------------------------------------------------------------------------
__global__ __launch_bounds__(64) void pool_part(
    const float* __restrict__ w, const _Float16* __restrict__ X,
    float* __restrict__ part) {
  const int d = threadIdx.x;   // 0..63
  const int h = blockIdx.x;    // 0..15
  const int b = blockIdx.y;    // 0..7
  const int c = blockIdx.z;    // 0..TCH-1
  const int t0 = c * (SS / TCH);
  const float*    wb = w + ((long)b * SS + t0) * NH + h;
  const _Float16* xb = X + ((long)b * SS + t0) * DTOK + h * DH + d;
  float acc = 0.0f;
#pragma unroll 8
  for (int t = 0; t < SS / TCH; ++t)
    acc += wb[(long)t * NH] * (float)xb[(long)t * DTOK];
  part[(long)c * GVEC + (long)b * DTOK + h * DH + d] = acc;
}

// Pooling, stage 2: outg[o] = (mul?mul[o]:1) * sum_c part[c][o]
__global__ __launch_bounds__(256) void pool_finish(
    const float* __restrict__ part, const float* __restrict__ mul,
    float* __restrict__ outg) {
  const int o = blockIdx.x * 256 + threadIdx.x;  // 0..GVEC-1
  float s = 0.0f;
#pragma unroll
  for (int c = 0; c < TCH; ++c) s += part[(long)c * GVEC + o];
  outg[o] = mul ? s * mul[o] : s;
}

// wk2[b][h][d] = wk[h][d] * qg[b][d]
__global__ __launch_bounds__(256) void fold_small(
    const float* __restrict__ wk, const float* __restrict__ qg,
    _Float16* __restrict__ outw) {
  const int i = blockIdx.x * 256 + threadIdx.x;  // h*1024 + d
  const int b = blockIdx.y;
  const int d = i & (DTOK - 1);
  outw[(long)b * NH * DTOK + i] = (_Float16)(wk[i] * qg[b * DTOK + d]);
}

// ow2[b][do][di] = ow[do][di] * kg[b][di]
__global__ __launch_bounds__(256) void fold_big(
    const float* __restrict__ ow, const float* __restrict__ kg,
    _Float16* __restrict__ outw) {
  const long i = (long)blockIdx.x * 256 + threadIdx.x;  // do*1024 + di
  const int  b = blockIdx.y;
  const int di = (int)(i & (DTOK - 1));
  outw[(long)b * DTOK * DTOK + i] = (_Float16)(ow[i] * kg[b * DTOK + di]);
}

// ---------------------------------------------------------------------------
static inline char* ws_take(char*& p, size_t bytes) {
  char* r = p;
  p += (bytes + 255) & ~(size_t)255;
  return r;
}

extern "C" void kernel_launch(void* const* d_in, const int* in_sizes, int n_in,
                              void* d_out, int out_size, void* d_ws,
                              size_t ws_size, hipStream_t stream) {
  (void)in_sizes; (void)n_in; (void)out_size; (void)ws_size;
  const float* x_q   = (const float*)d_in[0];
  const float* x_kv  = (const float*)d_in[1];
  const float* q_w   = (const float*)d_in[2];
  const float* k_w   = (const float*)d_in[3];
  const float* v_w   = (const float*)d_in[4];
  const float* wq_w  = (const float*)d_in[5];
  const float* wk_w  = (const float*)d_in[6];
  const float* out_w = (const float*)d_in[7];
  const float* q_b   = (const float*)d_in[8];
  const float* k_b   = (const float*)d_in[9];
  const float* v_b   = (const float*)d_in[10];
  const float* wq_b  = (const float*)d_in[11];
  const float* wk_b  = (const float*)d_in[12];
  const float* out_b = (const float*)d_in[13];
  float* outp = (float*)d_out;

  // ---- workspace layout (bump allocator, 256B aligned) ----
  char* p = (char*)d_ws;
  _Float16* xq_h  = (_Float16*)ws_take(p, (size_t)MM * DTOK * 2);
  _Float16* xkv_h = (_Float16*)ws_take(p, (size_t)MM * DTOK * 2);
  _Float16* qw_h  = (_Float16*)ws_take(p, (size_t)DTOK * DTOK * 2);
  _Float16* kw_h  = (_Float16*)ws_take(p, (size_t)DTOK * DTOK * 2);
  _Float16* vw_h  = (_Float16*)ws_take(p, (size_t)DTOK * DTOK * 2);
  _Float16* wq_h  = (_Float16*)ws_take(p, (size_t)NH * DTOK * 2);
  _Float16* q_h   = (_Float16*)ws_take(p, (size_t)MM * DTOK * 2);
  _Float16* k_h   = (_Float16*)ws_take(p, (size_t)MM * DTOK * 2);
  _Float16* v_h   = (_Float16*)ws_take(p, (size_t)MM * DTOK * 2);
  float*    alpha = (float*)   ws_take(p, (size_t)MM * NH * 4);
  float*    beta  = (float*)   ws_take(p, (size_t)MM * NH * 4);
  float*    part  = (float*)   ws_take(p, (size_t)TCH * GVEC * 4);
  float*    qg    = (float*)   ws_take(p, (size_t)GVEC * 4);
  float*    kg    = (float*)   ws_take(p, (size_t)GVEC * 4);
  _Float16* wk2   = (_Float16*)ws_take(p, (size_t)BB * NH * DTOK * 2);
  _Float16* ow2   = (_Float16*)ws_take(p, (size_t)BB * DTOK * DTOK * 2);

  const float inv_sqrt_dh = 0.125f;  // 1/sqrt(64)

  // ---- stage 1: f32 -> f16 conversions ----
  {
    const long nx = (long)MM * DTOK / 4;
    const long nw = (long)DTOK * DTOK / 4;
    const long ns = (long)NH * DTOK / 4;
    cvt_f32_to_f16<<<dim3((nx + 255) / 256), 256, 0, stream>>>(x_q, xq_h, nx);
    cvt_f32_to_f16<<<dim3((nx + 255) / 256), 256, 0, stream>>>(x_kv, xkv_h, nx);
    cvt_f32_to_f16<<<dim3((nw + 255) / 256), 256, 0, stream>>>(q_w, qw_h, nw);
    cvt_f32_to_f16<<<dim3((nw + 255) / 256), 256, 0, stream>>>(k_w, kw_h, nw);
    cvt_f32_to_f16<<<dim3((nw + 255) / 256), 256, 0, stream>>>(v_w, vw_h, nw);
    cvt_f32_to_f16<<<dim3((ns + 255) / 256), 256, 0, stream>>>(wq_w, wq_h, ns);
  }

  // ---- stage 2: q/k/v projections (32x64 wave tiles, pipelined) ----
  {
    dim3 g(MM / 128, DTOK / 128, 1);  // (256, 8)
    gemm_wmma_f16<2, 4, 4, 2, true, false, false, true><<<g, 256, 0, stream>>>(
        xq_h, qw_h, q_b, nullptr, nullptr, q_h, DTOK, DTOK, 0, 0, 0, 1.0f);
    gemm_wmma_f16<2, 4, 4, 2, true, false, false, true><<<g, 256, 0, stream>>>(
        xkv_h, kw_h, k_b, nullptr, nullptr, k_h, DTOK, DTOK, 0, 0, 0, 1.0f);
    gemm_wmma_f16<2, 4, 4, 2, true, false, false, true><<<g, 256, 0, stream>>>(
        xkv_h, vw_h, v_b, nullptr, nullptr, v_h, DTOK, DTOK, 0, 0, 0, 1.0f);
  }

  // ---- stage 3: alpha = (q @ wq^T + wq_b)/8; softmax; q_global ----
  gemm_wmma_f16<2, 1, 8, 1, true, false, true, false>
      <<<dim3(MM / 256, 1, 1), 256, 0, stream>>>(
          q_h, wq_h, wq_b, nullptr, alpha, nullptr, DTOK, NH, 0, 0, 0,
          inv_sqrt_dh);
  softmax_tokens<<<dim3(BB), 256, 0, stream>>>(alpha, SS);
  pool_part<<<dim3(NH, BB, TCH), 64, 0, stream>>>(alpha, q_h, part);
  pool_finish<<<dim3(GVEC / 256), 256, 0, stream>>>(part, nullptr, qg);

  // ---- stage 4: beta = (k @ (wk*qg)^T + wk_b)/8; softmax; k_global ----
  fold_small<<<dim3(NH * DTOK / 256, BB), 256, 0, stream>>>(wk_w, qg, wk2);
  gemm_wmma_f16<2, 1, 8, 1, true, false, true, false>
      <<<dim3(SS / 256, 1, BB), 256, 0, stream>>>(
          k_h, wk2, wk_b, nullptr, beta, nullptr, DTOK, NH,
          (long)SS * DTOK, (long)NH * DTOK, (long)SS * NH, inv_sqrt_dh);
  softmax_tokens<<<dim3(BB), 256, 0, stream>>>(beta, SS);
  pool_part<<<dim3(NH, BB, TCH), 64, 0, stream>>>(beta, k_h, part);
  pool_finish<<<dim3(GVEC / 256), 256, 0, stream>>>(part, qg, kg);

  // ---- stage 5: out = q + v @ (out_w*kg)^T + out_b ----
  fold_big<<<dim3(DTOK * DTOK / 256, BB), 256, 0, stream>>>(out_w, kg, ow2);
  gemm_wmma_f16<2, 4, 4, 2, true, true, true, false>
      <<<dim3(SS / 128, DTOK / 128, BB), 256, 0, stream>>>(
          v_h, ow2, out_b, q_h, outp, nullptr, DTOK, DTOK,
          (long)SS * DTOK, (long)DTOK * DTOK, (long)SS * DTOK, 1.0f);
}
